// ResTransformerBlock_75771813036524
// MI455X (gfx1250) — matrix operand order repaired and checked by reading, not compiled, not run on hardware
//
#include <hip/hip_runtime.h>
#include <math.h>

// ---------------------------------------------------------------------------
// ResTransformerBlock for MI455X (gfx1250, wave32, WMMA).
// conv: WMMA projections (weights DMA'd to LDS via global_load_async_to_lds,
// then swizzled to B-fragment order -> 2x ds_load_b128 per fragment) ->
// wave-per-edge alpha + atomic segment-max -> wave-per-edge exp + segment-sum
// + coalesced scatter -> node finalize (normalize + skip + BN stats) ->
// BN apply (+ final residual).
// ---------------------------------------------------------------------------

typedef __attribute__((ext_vector_type(16))) __bf16 v16bf;
typedef __attribute__((ext_vector_type(8)))  __bf16 v8bf;
typedef __attribute__((ext_vector_type(8)))  float  v8f;
typedef __attribute__((ext_vector_type(4)))  int    v4i;
typedef __attribute__((address_space(1))) v4i   gv4i;   // global-AS v4i
typedef __attribute__((address_space(3))) v4i   lv4i;   // LDS-AS v4i
typedef __attribute__((address_space(1))) float gfloat; // global-AS float

#define WAVES_PER_BLOCK 8

#if defined(__has_builtin)
#if __has_builtin(__builtin_amdgcn_global_load_async_to_lds_b128) && \
    __has_builtin(__builtin_amdgcn_s_wait_asynccnt)
#define HAVE_ASYNC_LDS 1
#endif
#endif
#ifndef HAVE_ASYNC_LDS
#define HAVE_ASYNC_LDS 0
#endif

// ------------------------------ utilities ---------------------------------
__global__ void zero_kernel(float* __restrict__ p, size_t n) {
    size_t t = (size_t)blockIdx.x * blockDim.x + threadIdx.x;
    size_t stride = (size_t)gridDim.x * blockDim.x;
    for (size_t i = t; i < n; i += stride) p[i] = 0.0f;
}

// Monotone order-preserving float<->uint key for atomicMax-based segment max.
__device__ __forceinline__ unsigned f32_ordered_key(float f) {
    unsigned u = __float_as_uint(f);
    return (u & 0x80000000u) ? ~u : (u | 0x80000000u);
}
__device__ __forceinline__ float key_to_f32(unsigned k) {
    return __uint_as_float((k & 0x80000000u) ? (k ^ 0x80000000u) : ~k);
}

// Force a global (AS1) store even when the pointer provenance is obscured,
// so codegen emits global_store_b32 instead of flat_store_b32.
__device__ __forceinline__ void store_global(float* p, size_t idx, float v) {
    ((gfloat*)p)[idx] = v;
}

// ------------------------- WMMA projection GEMM ---------------------------
// O_i = X @ W_i^T + b_i for 4 weight matrices [ODIM,KDIM], X [nrows,KDIM].
// Weights staged into LDS (async DMA), swizzled to bf16 B-fragment order:
//   Wf[(w*KSTEPS+s)*32 + l][16c+e] = W_w[16c+(l&15)][s*32+8*(l>>4)+(e&7)+16*(e>>3)]  (padded row ODIM+8)
// so a lane's fragment is two bank-spread ds_load_b128.
// One wave per (row-tile, weight-matrix) job for grid-level parallelism.
template<int KDIM, int ODIM>
__global__ __launch_bounds__(WAVES_PER_BLOCK * 32)
void proj_gemm_kernel(const float* __restrict__ X, int nrows,
                      const float* __restrict__ W0, const float* __restrict__ b0, float* __restrict__ O0,
                      const float* __restrict__ W1, const float* __restrict__ b1, float* __restrict__ O1,
                      const float* __restrict__ W2, const float* __restrict__ b2, float* __restrict__ O2,
                      const float* __restrict__ W3, const float* __restrict__ b3, float* __restrict__ O3)
{
    constexpr int KSTEPS = KDIM / 32;
    constexpr int WELEMS = ODIM * KDIM;            // per-matrix weight elements
    constexpr int LROW   = ODIM + 8;               // padded lane-row (bank spread)
    constexpr int NFRAGROWS = 4 * KSTEPS * 32;     // (w,s,lane) rows

    __shared__ __bf16 Wf[NFRAGROWS * LROW];        // conv1: 34 KB, conv2: 40 KB
    __shared__ float  Blds[4 * ODIM];

    const float* Wm[4] = {W0, W1, W2, W3};
    const float* Bm[4] = {b0, b1, b2, b3};
    float*       Om[4] = {O0, O1, O2, O3};

#if HAVE_ASYNC_LDS
    // CDNA5 async DMA: raw f32 weights global -> LDS, then swizzle LDS->LDS.
    __shared__ float Wraw[4 * WELEMS];             // 64 KB raw staging
    {
        constexpr int NCHUNK = 4 * WELEMS / 4;     // 16B chunks
        for (int i = threadIdx.x; i < NCHUNK; i += WAVES_PER_BLOCK * 32) {
            const int w   = i / (WELEMS / 4);
            const int rem = i - w * (WELEMS / 4);
            __builtin_amdgcn_global_load_async_to_lds_b128(
                (gv4i*)(Wm[w] + rem * 4),
                (lv4i*)(&Wraw[w * WELEMS + rem * 4]),
                0, 0);
        }
        __builtin_amdgcn_s_wait_asynccnt(0);
    }
    __syncthreads();
#endif

    // Swizzle to fragment order (bf16).
    for (int i = threadIdx.x; i < 4 * KSTEPS * 32 * ODIM; i += WAVES_PER_BLOCK * 32) {
        const int n   = i % ODIM;
        const int l   = (i / ODIM) & 31;
        const int wsi = i / (ODIM * 32);           // = w*KSTEPS + s
        const int w   = wsi / KSTEPS;
        const int s   = wsi - w * KSTEPS;
#if HAVE_ASYNC_LDS
        const float v = Wraw[w * WELEMS + ((n & ~15) + (l & 15)) * KDIM + s * 32 + (l >> 4) * 8 + (n & 7) + ((n >> 3) & 1) * 16];
#else
        const float v = Wm[w][(size_t)((n & ~15) + (l & 15)) * KDIM + s * 32 + (l >> 4) * 8 + (n & 7) + ((n >> 3) & 1) * 16];
#endif
        Wf[wsi * 32 * LROW + l * LROW + n] = (__bf16)v;
    }
    for (int i = threadIdx.x; i < 4 * ODIM; i += WAVES_PER_BLOCK * 32) {
        const int w = i / ODIM;
        Blds[i] = Bm[w][i - w * ODIM];
    }
    __syncthreads();

    const int lane = threadIdx.x & 31;
    const int wave = threadIdx.x >> 5;
    const int ntiles = (nrows + 15) >> 4;
    const int job  = blockIdx.x * WAVES_PER_BLOCK + wave;   // (tile, w) pair
    const int w    = job / ntiles;
    const int tile = job - w * ntiles;
    if (w < 4) {                                   // wave-uniform; EXEC all-ones inside
        const int half = lane >> 4;
        const int mrow = tile * 16 + (lane & 15);
        const int mclamped = mrow < nrows ? mrow : nrows - 1;
        const bool full = (tile * 16 + 16) <= nrows;   // wave-uniform fast path

        // A fragment (16x32 bf16 per k-step), ISA 7.12.2 layout:
        // lanes 0-15: K={0..7,16..23}; lanes 16-31: K={8..15,24..31} of row (lane&15).
        v16bf afrag[KSTEPS];
        {
            const float4* xr4 = (const float4*)(X + (size_t)mclamped * KDIM);
            #pragma unroll
            for (int s = 0; s < KSTEPS; ++s) {
                const float4 a0 = xr4[s * 8 + half * 2 + 0];      // K = half*8   .. +3
                const float4 a1 = xr4[s * 8 + half * 2 + 1];      // K = half*8+4 .. +7
                const float4 a2 = xr4[s * 8 + 4 + half * 2 + 0];  // K = 16+half*8   ..
                const float4 a3 = xr4[s * 8 + 4 + half * 2 + 1];  // K = 16+half*8+4 ..
                afrag[s][0]  = (__bf16)a0.x; afrag[s][1]  = (__bf16)a0.y;
                afrag[s][2]  = (__bf16)a0.z; afrag[s][3]  = (__bf16)a0.w;
                afrag[s][4]  = (__bf16)a1.x; afrag[s][5]  = (__bf16)a1.y;
                afrag[s][6]  = (__bf16)a1.z; afrag[s][7]  = (__bf16)a1.w;
                afrag[s][8]  = (__bf16)a2.x; afrag[s][9]  = (__bf16)a2.y;
                afrag[s][10] = (__bf16)a2.z; afrag[s][11] = (__bf16)a2.w;
                afrag[s][12] = (__bf16)a3.x; afrag[s][13] = (__bf16)a3.y;
                afrag[s][14] = (__bf16)a3.z; afrag[s][15] = (__bf16)a3.w;
            }
        }

        #pragma unroll
        for (int ct = 0; ct < ODIM / 16; ++ct) {
            v8f c = {};
            #pragma unroll
            for (int s = 0; s < KSTEPS; ++s) {
                // Lane l's fragment: 16 contiguous bf16 (two b128 LDS loads).
                const __bf16* base =
                    &Wf[((w * KSTEPS + s) * 32 + lane) * LROW + ct * 16];
                const v8bf lo = *(const v8bf*)(base);
                const v8bf hi = *(const v8bf*)(base + 8);
                const v16bf bfrag = __builtin_shufflevector(
                    lo, hi, 0, 1, 2, 3, 4, 5, 6, 7, 8, 9, 10, 11, 12, 13, 14, 15);
                c = __builtin_amdgcn_wmma_f32_16x16x32_bf16(
                        false, afrag[s], false, bfrag, (short)0, c, false, false);
            }
            // D layout: VGPR r, lane l -> row r + (l>>4)*8, col l&15.
            const int col  = ct * 16 + (lane & 15);
            const float bias = Blds[w * ODIM + col];
            float* O = Om[w];
            const int rowbase = tile * 16 + half * 8;
            if (full) {
                #pragma unroll
                for (int r = 0; r < 8; ++r)
                    store_global(O, (size_t)(rowbase + r) * ODIM + col, c[r] + bias);
            } else {
                #pragma unroll
                for (int r = 0; r < 8; ++r)
                    if (rowbase + r < nrows)
                        store_global(O, (size_t)(rowbase + r) * ODIM + col, c[r] + bias);
            }
        }
    }
}

// ---------------------------- edge phase ----------------------------------
// Wave-per-edge: lanes own 32 consecutive channels of one head -> fully
// coalesced 128B gathers, shuffle-reduced dot, coalesced (same-line) atomics.
template<int HEADS, int DH>   // DH == 32
__global__ void edge_alpha_kernel(const long long* __restrict__ ei, int E,
                                  const float* __restrict__ Q, const float* __restrict__ K,
                                  float* __restrict__ alpha, unsigned* __restrict__ mkey,
                                  float scale)
{
    const int lane = threadIdx.x & 31;
    const int wv = (int)((blockIdx.x * (size_t)blockDim.x + threadIdx.x) >> 5);
    if (wv >= E) return;                          // wave-uniform
    if (wv + 4096 < E) __builtin_prefetch(&ei[wv + 4096], 0, 1);  // global_prefetch
    const int src = (int)ei[wv];
    const int dst = (int)ei[(size_t)E + wv];
    const float* qrow = Q + (size_t)dst * (HEADS * DH);
    const float* krow = K + (size_t)src * (HEADS * DH);
    #pragma unroll
    for (int h = 0; h < HEADS; ++h) {
        float p = qrow[h * DH + lane] * krow[h * DH + lane];
        #pragma unroll
        for (int off = 16; off > 0; off >>= 1)
            p += __shfl_xor(p, off, 32);
        if (lane == 0) {
            const float a = p * scale;
            alpha[(size_t)wv * HEADS + h] = a;
            atomicMax(&mkey[(size_t)dst * HEADS + h], f32_ordered_key(a));
        }
    }
}

template<int HEADS, int DH>   // DH == 32
__global__ void edge_scatter_kernel(const long long* __restrict__ ei, int E,
                                    const float* __restrict__ V, const float* __restrict__ alpha,
                                    const unsigned* __restrict__ mkey,
                                    float* __restrict__ ssum, float* __restrict__ acc)
{
    const int lane = threadIdx.x & 31;
    const int wv = (int)((blockIdx.x * (size_t)blockDim.x + threadIdx.x) >> 5);
    if (wv >= E) return;
    if (wv + 4096 < E) __builtin_prefetch(&ei[wv + 4096], 0, 1);  // global_prefetch
    const int src = (int)ei[wv];
    const int dst = (int)ei[(size_t)E + wv];
    const float* vrow = V + (size_t)src * (HEADS * DH);
    float* orow = acc + (size_t)dst * (HEADS * DH);
    #pragma unroll
    for (int h = 0; h < HEADS; ++h) {
        // broadcast loads: all lanes same address, L0-served
        const float a = __expf(alpha[(size_t)wv * HEADS + h]
                               - key_to_f32(mkey[(size_t)dst * HEADS + h]));
        if (lane == 0) atomicAdd(&ssum[(size_t)dst * HEADS + h], a);
        atomicAdd(&orow[h * DH + lane], a * vrow[h * DH + lane]);
    }
}

// --------------------- node finalize + BN statistics ----------------------
template<int CH, int HEADS>
__global__ void node_finalize_kernel(int n, const float* skip,
                                     const float* __restrict__ acc,
                                     const float* __restrict__ ssum,
                                     float* h, float* __restrict__ sum,
                                     float* __restrict__ sumsq)
{
    constexpr int DH = CH / HEADS;
    const int t = blockIdx.x * blockDim.x + threadIdx.x;
    const int total = gridDim.x * blockDim.x;
    const int c = t % CH;
    const int stride = total / CH;
    const int r0 = t / CH;
    float ls = 0.f, lq = 0.f;
    for (int r = r0; r < n; r += stride) {
        const size_t idx = (size_t)r * CH + c;
        const float s = ssum[(size_t)r * HEADS + c / DH];
        const float v = skip[idx] + acc[idx] / (s + 1e-16f);
        h[idx] = v;
        ls += v;
        lq += v * v;
    }
    atomicAdd(&sum[c], ls);
    atomicAdd(&sumsq[c], lq);
}

// -------------------- BN apply (+ ReLU, + final residual) -----------------
template<int CH, bool RESID>
__global__ void bn_apply_kernel(int n, float* h,
                                const float* __restrict__ sum, const float* __restrict__ sumsq,
                                const float* __restrict__ g, const float* __restrict__ be,
                                const float* __restrict__ x, float* __restrict__ out)
{
    const size_t t = (size_t)blockIdx.x * blockDim.x + threadIdx.x;
    if (t >= (size_t)n * CH) return;
    const int c = (int)(t % CH);
    const float inv_n = 1.0f / (float)n;
    const float mu = sum[c] * inv_n;
    const float var = sumsq[c] * inv_n - mu * mu;     // biased variance
    float y = g[c] * (h[t] - mu) * rsqrtf(var + 1e-5f) + be[c];
    y = fmaxf(y, 0.0f);
    if (RESID)
        out[t] = (y + x[t]) * 0.70710678118654752f;   // (h + x)/sqrt(2)
    else
        h[t] = y;                                     // in place for next conv
}

// ------------------------------- launcher ---------------------------------
extern "C" void kernel_launch(void* const* d_in, const int* in_sizes, int n_in,
                              void* d_out, int out_size, void* d_ws, size_t ws_size,
                              hipStream_t stream)
{
    const float*     x  = (const float*)d_in[0];
    const long long* ei = (const long long*)d_in[1];          // int64 edge_index [2,E]
    const float *W1q = (const float*)d_in[2],  *b1q = (const float*)d_in[3];
    const float *W1k = (const float*)d_in[4],  *b1k = (const float*)d_in[5];
    const float *W1v = (const float*)d_in[6],  *b1v = (const float*)d_in[7];
    const float *W1s = (const float*)d_in[8],  *b1s = (const float*)d_in[9];
    const float *g1  = (const float*)d_in[10], *be1 = (const float*)d_in[11];
    const float *W2q = (const float*)d_in[12], *b2q = (const float*)d_in[13];
    const float *W2k = (const float*)d_in[14], *b2k = (const float*)d_in[15];
    const float *W2v = (const float*)d_in[16], *b2v = (const float*)d_in[17];
    const float *W2s = (const float*)d_in[18], *b2s = (const float*)d_in[19];
    const float *g2  = (const float*)d_in[20], *be2 = (const float*)d_in[21];

    const int N = in_sizes[0] / 32;
    const int E = in_sizes[1] / 2;
    constexpr int C = 32, HID = 128, H = 4;

    float* ws = (float*)d_ws;
    size_t off = 0;
    auto take = [&](size_t nf) { float* p = ws + off; off += nf; return p; };

    float* Q1 = take((size_t)N * HID);
    float* K1 = take((size_t)N * HID);
    float* V1 = take((size_t)N * HID);
    float* H1 = take((size_t)N * HID);   // skip term -> h1 -> relu(bn(h1)), in place
    float* A1 = take((size_t)E * H);
    float* Q2 = take((size_t)N * C);
    float* K2 = take((size_t)N * C);
    float* V2 = take((size_t)N * C);
    float* H2 = take((size_t)N * C);     // skip term -> h2, in place
    float* A2 = take((size_t)E);
    // Contiguous accumulator span, zeroed once per call (deterministic).
    float* zbase = ws + off;
    float* ACC1 = take((size_t)N * HID);
    float* SS1  = take((size_t)N * H);
    float* MK1  = take((size_t)N * H);   // uint keys; 0u is <= every real key
    float* SUM1 = take(HID);
    float* SQ1  = take(HID);
    float* ACC2 = take((size_t)N * C);
    float* SS2  = take((size_t)N);
    float* MK2  = take((size_t)N);
    float* SUM2 = take(C);
    float* SQ2  = take(C);
    const size_t zlen = (size_t)((ws + off) - zbase);

    zero_kernel<<<2048, 256, 0, stream>>>(zbase, zlen);

    const int tiles   = (N + 15) / 16;
    const int gblocks = (tiles * 4 + WAVES_PER_BLOCK - 1) / WAVES_PER_BLOCK;  // (tile,w) jobs
    const int eblocks = (E + WAVES_PER_BLOCK - 1) / WAVES_PER_BLOCK;          // wave-per-edge
    const float sc = 0.17677669529663688f;   // 1/sqrt(32) for both convs (D=32)

    // ---- conv1: C -> H*D, 4 heads ----
    proj_gemm_kernel<32, 128><<<gblocks, WAVES_PER_BLOCK * 32, 0, stream>>>(
        x, N, W1q, b1q, Q1, W1k, b1k, K1, W1v, b1v, V1, W1s, b1s, H1);

    edge_alpha_kernel<4, 32><<<eblocks, WAVES_PER_BLOCK * 32, 0, stream>>>(
        ei, E, Q1, K1, A1, (unsigned*)MK1, sc);
    edge_scatter_kernel<4, 32><<<eblocks, WAVES_PER_BLOCK * 32, 0, stream>>>(
        ei, E, V1, A1, (const unsigned*)MK1, SS1, ACC1);
    node_finalize_kernel<128, 4><<<256, 256, 0, stream>>>(
        N, H1, ACC1, SS1, H1, SUM1, SQ1);
    bn_apply_kernel<128, false><<<(int)(((size_t)N * 128 + 255) / 256), 256, 0, stream>>>(
        N, H1, SUM1, SQ1, g1, be1, nullptr, nullptr);

    // ---- conv2: HID -> C, 1 head ----
    proj_gemm_kernel<128, 32><<<gblocks, WAVES_PER_BLOCK * 32, 0, stream>>>(
        H1, N, W2q, b2q, Q2, W2k, b2k, K2, W2v, b2v, V2, W2s, b2s, H2);

    edge_alpha_kernel<1, 32><<<eblocks, WAVES_PER_BLOCK * 32, 0, stream>>>(
        ei, E, Q2, K2, A2, (unsigned*)MK2, sc);
    edge_scatter_kernel<1, 32><<<eblocks, WAVES_PER_BLOCK * 32, 0, stream>>>(
        ei, E, V2, A2, (const unsigned*)MK2, SS2, ACC2);
    node_finalize_kernel<32, 1><<<256, 256, 0, stream>>>(
        N, H2, ACC2, SS2, H2, SUM2, SQ2);
    bn_apply_kernel<32, true><<<(int)(((size_t)N * 32 + 255) / 256), 256, 0, stream>>>(
        N, H2, SUM2, SQ2, g2, be2, x, (float*)d_out);

    (void)n_in; (void)out_size; (void)ws_size;
}
